// CustomConv2d_41747082117243
// MI455X (gfx1250) — 8-bit matrix operand order repaired and checked by reading, compiled, not run on hardware
//
#include <hip/hip_runtime.h>

#define NB    32
#define C_IN  128
#define K_OUT 256
#define H     56
#define W     56
#define HW    (H * W)
#define HP    58
#define WP    58
#define KG    (C_IN * 9)
#define MG    (NB * HW)
#define XPADBYTES (NB * HP * WP * C_IN)

typedef __attribute__((ext_vector_type(8))) int v8i;

#if defined(__has_builtin)
#if __has_builtin(__builtin_amdgcn_global_load_async_to_lds_b128) && \
    __has_builtin(__builtin_amdgcn_s_wait_asynccnt)
#define USE_ASYNC 1
#endif
#endif

#ifdef USE_ASYNC
typedef int v4i_ __attribute__((vector_size(16)));
typedef __attribute__((address_space(1))) v4i_ gv4i;
typedef __attribute__((address_space(3))) v4i_ lv4i;
#define ASYNC_CP16(dst_lds, src_glb) \
    __builtin_amdgcn_global_load_async_to_lds_b128((gv4i*)(src_glb), (lv4i*)(dst_lds), 0, 0)
#define WAIT_ASYNC(n) __builtin_amdgcn_s_wait_asynccnt(n)
#else
#define ASYNC_CP16(dst_lds, src_glb) \
    (*(uint4*)(dst_lds) = *(const uint4*)(src_glb))
#define WAIT_ASYNC(n)
#endif

__global__ __launch_bounds__(256) void quantize_x(const float* __restrict__ x,
                                                  signed char* __restrict__ x8) {
    int idx = blockIdx.x * blockDim.x + threadIdx.x;
    if (idx >= XPADBYTES) return;
    int c   = idx % C_IN;
    int rem = idx / C_IN;
    int wpix = rem % WP;  rem /= WP;
    int hpix = rem % HP;  rem /= HP;
    int n    = rem;
    float v = 0.0f;
    if (hpix >= 1 && hpix <= H && wpix >= 1 && wpix <= W)
        v = x[((n * C_IN + c) * H + (hpix - 1)) * W + (wpix - 1)];
    x8[idx] = (signed char)__float2int_rn(v);
}

__global__ __launch_bounds__(256) void pack_w(const float* __restrict__ wq,
                                              unsigned int* __restrict__ wp) {
    int idx = blockIdx.x * blockDim.x + threadIdx.x;
    if (idx >= (KG / 4) * K_OUT) return;
    int ko = idx % K_OUT;
    int kq = idx / K_OUT;
    int k  = kq * 4;
    int rs = k / C_IN;
    int c0 = k % C_IN;
    int r = rs / 3, s = rs % 3;
    unsigned int dw = 0;
#pragma unroll
    for (int b = 0; b < 4; ++b) {
        float v = wq[((ko * C_IN + (c0 + b)) * 3 + r) * 3 + s];
        int   q = __float2int_rn(v);
        dw |= ((unsigned int)(unsigned char)(signed char)q) << (8 * b);
    }
    wp[kq * K_OUT + ko] = dw;
}

__device__ __forceinline__ float requant(float f) {
    f = rintf(f * (1.0f / 128.0f));
    f = fminf(fmaxf(f, -128.0f), 127.0f);
    return fmaxf(f, 0.0f);
}

__global__ __launch_bounds__(256) void qconv_wmma(const signed char* __restrict__ x8,
                                                  const unsigned int* __restrict__ wp,
                                                  const float* __restrict__ bias,
                                                  float* __restrict__ out) {
    __shared__ __align__(16) unsigned char smem[36864];

    const int t    = threadIdx.x;
    const int wave = t >> 5;
    const int lane = t & 31;
    const int half = lane >> 4;
    const int l16  = lane & 15;

    const int block_m = blockIdx.x * 64;
    const int block_n = blockIdx.y * 128;
    const int wave_m  = (wave >> 2) * 32;
    const int wave_n  = (wave & 3) * 32;

    const int arow = t >> 2;
    const int acol = (t & 3) * 16;
    const int mA   = block_m + arow;
    const int nA   = mA / HW;
    const int pqA  = mA % HW;
    const int pA   = pqA / W, qA = pqA % W;
    const int aBase = ((nA * HP + pA) * WP + qA) * C_IN;

    const int d0 = t * 4;
    const int d1 = t * 4 + 1024;

    v8i acc00 = {}, acc01 = {}, acc10 = {}, acc11 = {};

    auto issue = [&](int ks, int buf) {
        unsigned char* Adst = smem + buf * 4096;
        unsigned char* Bdst = smem + 8192 + buf * 8192;
        const int rs = ks >> 1;
        const int r  = rs / 3, s = rs - r * 3;
        const int c0 = (ks & 1) * 64;
        ASYNC_CP16(Adst + arow * 64 + acol,
                   x8 + aBase + (r * WP + s) * C_IN + c0 + acol);
        const int kq0 = ks * 16;
        ASYNC_CP16(Bdst + d0 * 4,
                   wp + (kq0 + (d0 >> 7)) * K_OUT + block_n + (d0 & 127));
        ASYNC_CP16(Bdst + d1 * 4,
                   wp + (kq0 + (d1 >> 7)) * K_OUT + block_n + (d1 & 127));
    };

    issue(0, 0);
    for (int ks = 0; ks < KG / 64; ++ks) {
        const int cur = ks & 1;
        if (ks < (KG / 64 - 1)) {
            issue(ks + 1, cur ^ 1);
            WAIT_ASYNC(3);
        } else {
            WAIT_ASYNC(0);
        }
        __syncthreads();

        const unsigned char* Ab = smem + cur * 4096;
        const unsigned int*  Bb = (const unsigned int*)(smem + 8192 + cur * 8192);
        v8i a0, a1, b0, b1;
#pragma unroll
        for (int j = 0; j < 8; ++j) {
            const int ka = (j >> 2) * 32 + half * 16 + (j & 3) * 4;
            a0[j] = *(const int*)(Ab + (wave_m + l16) * 64 + ka);
            a1[j] = *(const int*)(Ab + (wave_m + 16 + l16) * 64 + ka);
            const int kb = (j >> 2) * 32 + half * 16 + (j & 3) * 4;
            b0[j] = (int)Bb[(kb >> 2) * 128 + wave_n + l16];
            b1[j] = (int)Bb[(kb >> 2) * 128 + wave_n + 16 + l16];
        }
        acc00 = __builtin_amdgcn_wmma_i32_16x16x64_iu8(true, a0, true, b0, acc00, false, false);
        acc01 = __builtin_amdgcn_wmma_i32_16x16x64_iu8(true, a0, true, b1, acc01, false, false);
        acc10 = __builtin_amdgcn_wmma_i32_16x16x64_iu8(true, a1, true, b0, acc10, false, false);
        acc11 = __builtin_amdgcn_wmma_i32_16x16x64_iu8(true, a1, true, b1, acc11, false, false);
        __syncthreads();
    }

    float* T = (float*)(smem + wave * 4608);
    const float bv0 = bias[block_n + wave_n + l16];
    const float bv1 = bias[block_n + wave_n + 16 + l16];
#pragma unroll
    for (int v = 0; v < 8; ++v) {
        const int m = v + half * 8;
        T[(l16)      * 36 + m]      = requant((float)acc00[v] + bv0);
        T[(16 + l16) * 36 + m]      = requant((float)acc01[v] + bv1);
        T[(l16)      * 36 + 16 + m] = requant((float)acc10[v] + bv0);
        T[(16 + l16) * 36 + 16 + m] = requant((float)acc11[v] + bv1);
    }
    const int n_g    = block_n + wave_n + lane;
    const int m_base = block_m + wave_m;
    const int nb  = m_base / HW;
    const int pq0 = m_base % HW;
    float4*       op = (float4*)(out + (size_t)(nb * K_OUT + n_g) * HW + pq0);
    const float4* Tr = (const float4*)(T + lane * 36);
#pragma unroll
    for (int i = 0; i < 8; ++i) op[i] = Tr[i];
}

extern "C" void kernel_launch(void* const* d_in, const int* in_sizes, int n_in,
                              void* d_out, int out_size, void* d_ws, size_t ws_size,
                              hipStream_t stream) {
    const float* x_q  = (const float*)d_in[0];
    const float* w_q  = (const float*)d_in[1];
    const float* bias = (const float*)d_in[2];
    float* out = (float*)d_out;

    signed char*  x8  = (signed char*)d_ws;
    unsigned int* wpk = (unsigned int*)((char*)d_ws + XPADBYTES);

    quantize_x<<<(XPADBYTES + 255) / 256, 256, 0, stream>>>(x_q, x8);
    pack_w<<<((KG / 4) * K_OUT + 255) / 256, 256, 0, stream>>>(w_q, wpk);

    dim3 grid(MG / 64, K_OUT / 128);
    qconv_wmma<<<grid, 256, 0, stream>>>(x8, wpk, bias, out);
}
